// LlamaAttention_31258771980331
// MI455X (gfx1250) — matrix operand order repaired and checked by reading, not compiled, not run on hardware
//
#include <hip/hip_runtime.h>

typedef __attribute__((ext_vector_type(16))) _Float16 v16h;
typedef __attribute__((ext_vector_type(8)))  _Float16 v8h;
typedef __attribute__((ext_vector_type(8)))  float    v8f;
typedef __attribute__((ext_vector_type(4)))  int      v4i;

#define HIDDEN 2048
#define SEQ    2048
#define BATCH  2
#define HEADS  16
#define HDIM   128

// ---------------------------------------------------------------------------
// CDNA5 async memory->LDS copies (GLOBAL_LOAD_ASYNC_TO_LDS_B128, ASYNCcnt)
// Builtin signature (from toolchain diagnostic): (v4i as1*, v4i as3*, Ii, Ii)
// ---------------------------------------------------------------------------
#if defined(__has_builtin)
#  if __has_builtin(__builtin_amdgcn_global_load_async_to_lds_b128)
#    define HAVE_ASYNC 1
#  endif
#endif
#ifndef HAVE_ASYNC
#  define HAVE_ASYNC 0
#endif

#if HAVE_ASYNC
#  if __has_builtin(__builtin_amdgcn_s_wait_asynccnt)
#    define WAIT_ASYNC(n) __builtin_amdgcn_s_wait_asynccnt(n)
#  else
#    define WAIT_ASYNC(n) asm volatile("s_wait_asynccnt %0" ::"i"(n) : "memory")
#  endif
#else
#  define WAIT_ASYNC(n) ((void)0)
#endif

typedef __attribute__((address_space(1))) v4i gv4i;  // global 128-bit chunk
typedef __attribute__((address_space(3))) v4i lv4i;  // LDS 128-bit chunk

// copy 16 bytes (8 f16) global -> LDS
__device__ __forceinline__ void async_cp16(const _Float16* g, _Float16* l) {
#if HAVE_ASYNC
  __builtin_amdgcn_global_load_async_to_lds_b128((gv4i*)g, (lv4i*)l, 0, 0);
#else
  *(v8h*)l = *(const v8h*)g;
#endif
}

// ---------------------------------------------------------------------------
// WMMA fragment loaders (layouts per CDNA5 ISA 7.12.2, wave32)
// A (16x32 f16): lane holds row M=lane%16; halves 0..7 -> K=koff..koff+7,
//                halves 8..15 -> K=koff+16..koff+23, koff = 8*(lane>=16)
// B (32x16 f16): lane holds col N=lane%16; halves 0..7 -> K=koff..koff+7,
//                halves 8..15 -> K=koff+16..koff+23, koff = 8*(lane>=16)
// C (16x16 f32): lane col N=lane%16; VGPR i -> row M = i + 8*(lane>=16)
// ---------------------------------------------------------------------------
__device__ __forceinline__ v16h frag_a_ld(const _Float16* base, int ldr, int lane) {
  const _Float16* p = base + (size_t)(lane & 15) * ldr + ((lane >> 4) << 3);
  v8h lo = *(const v8h*)(p);
  v8h hi = *(const v8h*)(p + 16);
  return __builtin_shufflevector(lo, hi, 0,1,2,3,4,5,6,7,8,9,10,11,12,13,14,15);
}
__device__ __forceinline__ v16h frag_b_ld(const _Float16* base, int ldr, int lane) {
  const _Float16* p = base + (size_t)(lane & 15) * ldr + ((lane >> 4) << 3);
  v8h lo = *(const v8h*)(p);
  v8h hi = *(const v8h*)(p + 16);
  return __builtin_shufflevector(lo, hi, 0,1,2,3,4,5,6,7,8,9,10,11,12,13,14,15);
}

__device__ __forceinline__ float hmax16(float v) {
  v = fmaxf(v, __shfl_xor(v, 1, 32));
  v = fmaxf(v, __shfl_xor(v, 2, 32));
  v = fmaxf(v, __shfl_xor(v, 4, 32));
  v = fmaxf(v, __shfl_xor(v, 8, 32));
  return v;
}
__device__ __forceinline__ float hsum16(float v) {
  v += __shfl_xor(v, 1, 32);
  v += __shfl_xor(v, 2, 32);
  v += __shfl_xor(v, 4, 32);
  v += __shfl_xor(v, 8, 32);
  return v;
}

// ---------------------------------------------------------------------------
// elementwise fp32 -> f16
// ---------------------------------------------------------------------------
__global__ __launch_bounds__(256) void f32_to_f16_kernel(const float* __restrict__ in,
                                                         _Float16* __restrict__ out,
                                                         int n) {
  int i = blockIdx.x * 256 + threadIdx.x;
  if (i < n) out[i] = (_Float16)in[i];
}

// ---------------------------------------------------------------------------
// W (2048x2048 f32, row-major) -> Wt (2048x2048 f16, row-major) = W^T
// ---------------------------------------------------------------------------
__global__ __launch_bounds__(256) void transpose_f32_f16_kernel(const float* __restrict__ W,
                                                                _Float16* __restrict__ Wt) {
  __shared__ float tile[32][33];
  const int tx = threadIdx.x & 31, ty = threadIdx.x >> 5;
  const int k0 = blockIdx.y * 32, n0 = blockIdx.x * 32;
#pragma unroll
  for (int i = 0; i < 4; ++i)
    tile[ty + 8 * i][tx] = W[(size_t)(k0 + ty + 8 * i) * HIDDEN + n0 + tx];
  __syncthreads();
#pragma unroll
  for (int i = 0; i < 4; ++i)
    Wt[(size_t)(n0 + ty + 8 * i) * HIDDEN + k0 + tx] = (_Float16)tile[tx][ty + 8 * i];
}

// ---------------------------------------------------------------------------
// C(M,N) = A(MxK) * Bt(NxK)^T, f16 in, f32 accumulate.
// MODE 0: store f16 row-major   (Q, K projections)
// MODE 1: store f16 as (b, n, s) — per-head transposed V for attention
// MODE 2: store f32 row-major   (final output)
// 256 threads = 8 waves; WG tile 128x128; wave tile 32x64; K-step 32.
// Double-buffered LDS fed by async global->LDS copies.
// ---------------------------------------------------------------------------
template <int MODE>
__global__ __launch_bounds__(256) void gemm_wmma_kernel(const _Float16* __restrict__ A,
                                                        const _Float16* __restrict__ Bt,
                                                        void* __restrict__ Out,
                                                        int M, int N, int Kd) {
  __shared__ _Float16 tA[2][128 * 40];
  __shared__ _Float16 tB[2][128 * 40];
  const int tid  = threadIdx.x;
  const int lane = tid & 31;
  const int wave = tid >> 5;
  const int mBase = blockIdx.y * 128;
  const int nBase = blockIdx.x * 128;
  const int wm = (wave & 3) * 32;
  const int wn = (wave >> 2) * 64;

  v8f acc[2][4] = {};

  auto stage = [&](int buf, int kk) {
#pragma unroll
    for (int i = 0; i < 2; ++i) {
      int idx = tid + 256 * i;
      int r = idx >> 2, seg = idx & 3;
      async_cp16(&A[(size_t)(mBase + r) * Kd + kk + seg * 8], &tA[buf][r * 40 + seg * 8]);
      async_cp16(&Bt[(size_t)(nBase + r) * Kd + kk + seg * 8], &tB[buf][r * 40 + seg * 8]);
    }
  };

  const int nsteps = Kd / 32;
  stage(0, 0);  // 4 async copies per thread in flight

  for (int it = 0; it < nsteps; ++it) {
    const int buf = it & 1;
    if (it + 1 < nsteps) {
      stage(buf ^ 1, (it + 1) * 32);  // 8 in flight; oldest 4 are current stage
      WAIT_ASYNC(4);
    } else {
      WAIT_ASYNC(0);
    }
    __syncthreads();

    v16h aF[2], bF[4];
#pragma unroll
    for (int t = 0; t < 2; ++t) aF[t] = frag_a_ld(&tA[buf][(wm + t * 16) * 40], 40, lane);
#pragma unroll
    for (int u = 0; u < 4; ++u) bF[u] = frag_b_ld(&tB[buf][(wn + u * 16) * 40], 40, lane);
#pragma unroll
    for (int t = 0; t < 2; ++t)
#pragma unroll
      for (int u = 0; u < 4; ++u)
        acc[t][u] = __builtin_amdgcn_wmma_f32_16x16x32_f16(
            false, aF[t], false, bF[u], (short)0, acc[t][u], false, false);
    __syncthreads();
  }

  const int hl = lane >> 4;
  const int ln = lane & 15;
#pragma unroll
  for (int t = 0; t < 2; ++t) {
#pragma unroll
    for (int u = 0; u < 4; ++u) {
      const int n = nBase + wn + u * 16 + ln;
#pragma unroll
      for (int i = 0; i < 8; ++i) {
        const int m = mBase + wm + t * 16 + i + 8 * hl;
        const float v = acc[t][u][i];
        if (MODE == 0) {
          ((_Float16*)Out)[(size_t)m * N + n] = (_Float16)v;
        } else if (MODE == 1) {
          const int b = m >> 11, s = m & (SEQ - 1);
          ((_Float16*)Out)[((size_t)(b * N + n)) * SEQ + s] = (_Float16)v;
        } else {
          ((float*)Out)[(size_t)m * N + n] = v;
        }
      }
    }
  }
}

// ---------------------------------------------------------------------------
// In-place RoPE on f16 activations laid out (B*S, H*D).
// out[d] = x[d]*cos(e[d]) + x[(d-1) mod 128]*sin(e[d]),  e[d] = pos * invf[d%64]
// One wave per 128-dim head vector; all lanes load before any lane stores.
// ---------------------------------------------------------------------------
__global__ __launch_bounds__(256) void rope_kernel(_Float16* __restrict__ X, int nvec) {
  const int g = blockIdx.x * 256 + threadIdx.x;
  const int lane = g & 31;
  const int vec = g >> 5;
  if (vec >= nvec) return;
  const int h = vec & (HEADS - 1);
  const int m = vec >> 4;
  const int pos = m & (SEQ - 1);
  _Float16* base = X + (size_t)m * HIDDEN + h * HDIM;

  const int d0 = lane * 4;
  float xv[5];
  xv[0] = (float)base[(d0 + 127) & 127];
#pragma unroll
  for (int dd = 0; dd < 4; ++dd) xv[dd + 1] = (float)base[d0 + dd];

  float out[4];
#pragma unroll
  for (int dd = 0; dd < 4; ++dd) {
    const int j = (d0 + dd) & 63;
    // 10000^(-j/64) = exp2(-log2(10000) * j / 64)
    const float freq = (float)pos * exp2f(-13.287712379549449f * (float)j * (1.0f / 64.0f));
    float sn, cs;
    sincosf(freq, &sn, &cs);
    out[dd] = xv[dd + 1] * cs + xv[dd] * sn;
  }
#pragma unroll
  for (int dd = 0; dd < 4; ++dd) base[d0 + dd] = (_Float16)out[dd];
}

// ---------------------------------------------------------------------------
// Flash-style attention. Grid (S/64, H, B); 128 threads = 4 waves;
// each wave owns 16 query rows, streams keys in chunks of 32 with
// double-buffered async K/V staging.
// Q: (B*S, 2048) f16 (roped).  K: same.  Vt: (b, h*128+d, s) f16.
// Ctx out: (B*S, 2048) f16.
// ---------------------------------------------------------------------------
__global__ __launch_bounds__(128) void attn_wmma_kernel(const _Float16* __restrict__ Q,
                                                        const _Float16* __restrict__ K,
                                                        const _Float16* __restrict__ Vt,
                                                        _Float16* __restrict__ Ctx) {
  __shared__ _Float16 kb[2][32 * 136];     // 32 keys x 128 d (+pad)
  __shared__ _Float16 vb[2][128 * 40];     // 128 d x 32 keys (+pad)
  __shared__ _Float16 pb[4 * 16 * 40];     // per-wave P staging

  const int tid = threadIdx.x, lane = tid & 31, wave = tid >> 5;
  const int h = blockIdx.y, b = blockIdx.z;
  const int qBase = blockIdx.x * 64 + wave * 16;
  const int hl = lane >> 4, ln = lane & 15;

  // Q fragments held in registers for the whole key sweep
  const _Float16* qrow = Q + ((size_t)(b * SEQ + qBase)) * HIDDEN + h * HDIM;
  v16h qF[4];
#pragma unroll
  for (int dc = 0; dc < 4; ++dc) qF[dc] = frag_a_ld(qrow + dc * 32, HIDDEN, lane);

  v8f cacc[8] = {};
  float mrow[8], lrow[8];
#pragma unroll
  for (int i = 0; i < 8; ++i) { mrow[i] = -1e30f; lrow[i] = 0.0f; }

  const _Float16* kbg = K + ((size_t)(b * SEQ)) * HIDDEN + h * HDIM;
  const _Float16* vbg = Vt + ((size_t)(b * HIDDEN + h * HDIM)) * SEQ;
  _Float16* pmy = &pb[wave * 16 * 40];
  const float scale = 0.08838834764831845f;  // 1/sqrt(128)

  auto stage = [&](int buf, int c0) {
#pragma unroll
    for (int i = 0; i < 4; ++i) {
      int idx = tid + 128 * i;
      int r = idx >> 4, seg = idx & 15;
      async_cp16(&kbg[(size_t)(c0 + r) * HIDDEN + seg * 8], &kb[buf][r * 136 + seg * 8]);
      int d = idx >> 2, sg = idx & 3;
      async_cp16(&vbg[(size_t)d * SEQ + c0 + sg * 8], &vb[buf][d * 40 + sg * 8]);
    }
  };

  const int nsteps = SEQ / 32;  // 64
  stage(0, 0);  // 8 async copies per thread in flight

  for (int it = 0; it < nsteps; ++it) {
    const int buf = it & 1;
    if (it + 1 < nsteps) {
      stage(buf ^ 1, (it + 1) * 32);  // 16 in flight; oldest 8 are current stage
      WAIT_ASYNC(8);
    } else {
      WAIT_ASYNC(0);
    }
    __syncthreads();

    // scores: (16 q) x (32 keys), accumulate over d=128
    v8f s0 = {}, s1 = {};
#pragma unroll
    for (int dc = 0; dc < 4; ++dc) {
      v16h k0 = frag_b_ld(&kb[buf][0 * 136 + dc * 32], 136, lane);
      v16h k1 = frag_b_ld(&kb[buf][16 * 136 + dc * 32], 136, lane);
      s0 = __builtin_amdgcn_wmma_f32_16x16x32_f16(false, qF[dc], false, k0, (short)0, s0, false, false);
      s1 = __builtin_amdgcn_wmma_f32_16x16x32_f16(false, qF[dc], false, k1, (short)0, s1, false, false);
    }

    // online softmax; C-fragment row i lives in VGPR i across a 16-lane half
    float alpha[8];
#pragma unroll
    for (int i = 0; i < 8; ++i) {
      const float v0 = s0[i] * scale;
      const float v1 = s1[i] * scale;
      const float cm = hmax16(fmaxf(v0, v1));
      const float mn = fmaxf(mrow[i], cm);
      const float al = __expf(mrow[i] - mn);
      const float p0 = __expf(v0 - mn);
      const float p1 = __expf(v1 - mn);
      const float rs = hsum16(p0 + p1);
      lrow[i] = lrow[i] * al + rs;
      mrow[i] = mn;
      alpha[i] = al;
      pmy[(i + 8 * hl) * 40 + ln]      = (_Float16)p0;
      pmy[(i + 8 * hl) * 40 + 16 + ln] = (_Float16)p1;
    }
#pragma unroll
    for (int dt = 0; dt < 8; ++dt)
#pragma unroll
      for (int i = 0; i < 8; ++i) cacc[dt][i] *= alpha[i];

    // P (16x32, A-frag via per-wave LDS round trip) @ V chunk (32 keys x 128 d)
    v16h pF = frag_a_ld(pmy, 40, lane);
#pragma unroll
    for (int dt = 0; dt < 8; ++dt) {
      v16h vF = frag_b_ld(&vb[buf][dt * 16 * 40], 40, lane);
      cacc[dt] = __builtin_amdgcn_wmma_f32_16x16x32_f16(false, pF, false, vF, (short)0, cacc[dt], false, false);
    }
    __syncthreads();
  }

  float invl[8];
#pragma unroll
  for (int i = 0; i < 8; ++i) invl[i] = 1.0f / lrow[i];
#pragma unroll
  for (int dt = 0; dt < 8; ++dt)
#pragma unroll
    for (int i = 0; i < 8; ++i) {
      const int qr = qBase + i + 8 * hl;
      Ctx[((size_t)(b * SEQ + qr)) * HIDDEN + h * HDIM + dt * 16 + ln] =
          (_Float16)(cacc[dt][i] * invl[i]);
    }
}

// ---------------------------------------------------------------------------
extern "C" void kernel_launch(void* const* d_in, const int* in_sizes, int n_in,
                              void* d_out, int out_size, void* d_ws, size_t ws_size,
                              hipStream_t stream) {
  const float* hidden = (const float*)d_in[0];
  const float* Wq = (const float*)d_in[1];
  const float* Wk = (const float*)d_in[2];
  const float* Wv = (const float*)d_in[3];
  const float* Wo = (const float*)d_in[4];

  const int M = BATCH * SEQ;               // 4096
  const size_t actBytes = (size_t)M * HIDDEN * sizeof(_Float16);       // 16 MB
  const size_t wBytes   = (size_t)HIDDEN * HIDDEN * sizeof(_Float16); //  8 MB

  char* ws = (char*)d_ws;
  size_t off = 0;
  _Float16* x16  = (_Float16*)(ws + off); off += actBytes;
  _Float16* wqT  = (_Float16*)(ws + off); off += wBytes;
  _Float16* wkT  = (_Float16*)(ws + off); off += wBytes;
  _Float16* wvT  = (_Float16*)(ws + off); off += wBytes;
  _Float16* woT  = (_Float16*)(ws + off); off += wBytes;
  _Float16* q16  = (_Float16*)(ws + off); off += actBytes;
  _Float16* k16  = (_Float16*)(ws + off); off += actBytes;
  _Float16* vt16 = (_Float16*)(ws + off); off += actBytes;
  _Float16* c16  = (_Float16*)(ws + off); off += actBytes;
  (void)ws_size; (void)in_sizes; (void)n_in; (void)out_size;

  // 1) convert activations, transpose+convert weights
  const int nAct = M * HIDDEN;
  f32_to_f16_kernel<<<(nAct + 255) / 256, 256, 0, stream>>>(hidden, x16, nAct);
  dim3 tg(HIDDEN / 32, HIDDEN / 32);
  transpose_f32_f16_kernel<<<tg, 256, 0, stream>>>(Wq, wqT);
  transpose_f32_f16_kernel<<<tg, 256, 0, stream>>>(Wk, wkT);
  transpose_f32_f16_kernel<<<tg, 256, 0, stream>>>(Wv, wvT);
  transpose_f32_f16_kernel<<<tg, 256, 0, stream>>>(Wo, woT);

  // 2) QKV projections (V stored per-head transposed for the P@V operand)
  dim3 gg(HIDDEN / 128, M / 128);
  gemm_wmma_kernel<0><<<gg, 256, 0, stream>>>(x16, wqT, q16, M, HIDDEN, HIDDEN);
  gemm_wmma_kernel<0><<<gg, 256, 0, stream>>>(x16, wkT, k16, M, HIDDEN, HIDDEN);
  gemm_wmma_kernel<1><<<gg, 256, 0, stream>>>(x16, wvT, vt16, M, HIDDEN, HIDDEN);

  // 3) RoPE in place on Q and K
  const int nvec = M * HEADS;              // 65536 head-vectors
  rope_kernel<<<(nvec * 32) / 256, 256, 0, stream>>>(q16, nvec);
  rope_kernel<<<(nvec * 32) / 256, 256, 0, stream>>>(k16, nvec);

  // 4) attention
  dim3 ag(SEQ / 64, HEADS, BATCH);
  attn_wmma_kernel<<<ag, 128, 0, stream>>>(q16, k16, vt16, c16);

  // 5) output projection -> f32 d_out
  gemm_wmma_kernel<2><<<gg, 256, 0, stream>>>(c16, woT, d_out, M, HIDDEN, HIDDEN);
}
